// MultiHeadEquivariant_65944927863261
// MI455X (gfx1250) — matrix operand order repaired and checked by reading, not compiled, not run on hardware
//
#include <hip/hip_runtime.h>

// ---------------------------------------------------------------------------
// Fused MultiHeadEquivariant for MI455X (gfx1250, wave32, WMMA).
//
// Per batch sample b (1024 of them), three chained 256x256x256 GEMMs:
//   H  = WbT  @ X           (WbT = w_before^T, rows permuted head-major)
//   Z  = H    @ Kh^T        (per-head group conv; kfull[h][k][g] precomputed)
//   Out= WaT  @ Z  + bias   (WaT = w_after^T with head-major column perm)
//
// One workgroup (256 threads = 8 wave32) per sample; H/Z live in LDS,
// x read once from HBM, out written once.  All matmuls via
// v_wmma_f32_16x16x32_bf16 (bf16 inputs, fp32 accumulate).
// bf16 conversions use native casts so the backend can emit hardware
// (packed) cvt instructions instead of manual RNE bit-twiddling.
// ---------------------------------------------------------------------------

#define FEAT  256
#define NSYM  256
#define NHEAD 8

typedef __attribute__((ext_vector_type(16))) __bf16       v16bf;
typedef __attribute__((ext_vector_type(2)))  __bf16       bf16x2;
typedef __attribute__((ext_vector_type(8)))  float        v8f;
typedef __attribute__((ext_vector_type(4)))  unsigned int u32x4;
typedef __attribute__((ext_vector_type(4)))  float        f32x4;

union Frag {            // 32 bytes: one WMMA bf16 A- or B-operand per lane
  v16bf v;
  u32x4 q[2];
};

__device__ __forceinline__ unsigned short f2bf(float f) {
  return __builtin_bit_cast(unsigned short, (__bf16)f);   // native cvt, RNE
}
__device__ __forceinline__ unsigned pack2bf(float lo, float hi) {
  bf16x2 p = { (__bf16)lo, (__bf16)hi };                  // packed cvt candidate
  return __builtin_bit_cast(unsigned, p);
}

constexpr int HS = 264;   // Hs row stride (bf16 elems): 256 + 8 pad, 16B-aligned rows,
constexpr int PS = 264;   // 264*2 = 528B = 132 words = 4 mod 64 banks -> conflict-free

// ---------------------------------------------------------------------------
// Prep: bf16-convert + permute/transposed weight tables into workspace.
//   kfB[(h*256+k)*256+g] = bf16(kernel[h, pt[k,g]])      (B-fragment layout)
//   wbT[o'*256+f]        = bf16(w_before[f, o(o')])       (A rows, head-major o')
//   waT[o2*256+o']       = bf16(w_after [o(o'), o2])      (A rows, o' columns)
// with o' = h*32+d  <->  o = d*8+h.
// ---------------------------------------------------------------------------
__global__ void __launch_bounds__(256)
prep_kernel(const float* __restrict__ wb, const float* __restrict__ kr,
            const float* __restrict__ wa, const int* __restrict__ pt,
            unsigned short* __restrict__ kfB,
            unsigned short* __restrict__ wbT,
            unsigned short* __restrict__ waT)
{
  const int idx = blockIdx.x * 256 + threadIdx.x;        // 0 .. 524287
  {
    const int h = idx >> 16, k = (idx >> 8) & 255, g = idx & 255;
    kfB[idx] = f2bf(kr[h * NSYM + pt[k * NSYM + g]]);
  }
  if (idx < FEAT * FEAT) {
    const int row = idx >> 8, col = idx & 255;
    { const int hh = row >> 5, dd = row & 31, o = dd * NHEAD + hh;
      wbT[idx] = f2bf(wb[col * FEAT + o]); }
    { const int hh = col >> 5, dd = col & 31, o = dd * NHEAD + hh;
      waT[idx] = f2bf(wa[o * FEAT + row]); }
  }
}

// ---------------------------------------------------------------------------
// Fused main kernel: one block per batch sample.
// ---------------------------------------------------------------------------
__global__ void __launch_bounds__(256)
fused_kernel(const float* __restrict__ x,
             const unsigned short* __restrict__ wbT,
             const unsigned short* __restrict__ kfB,
             const unsigned short* __restrict__ waT,
             const float* __restrict__ bias,
             float* __restrict__ out)
{
  __shared__ __align__(16) unsigned short Hs[FEAT * HS]; // H (o' x g), bf16, 132KB
  __shared__ __align__(16) unsigned short Xs[16 * PS];   // X panel, transposed [g][f]
  __shared__ __align__(16) unsigned short Zs[16 * PS];   // Z panel, transposed [k][o']

  const int tid  = threadIdx.x;
  const int lane = tid & 31;
  const int wave = tid >> 5;
  const int l15  = lane & 15;
  const int lh   = lane >> 4;           // 0: lanes 0-15, 1: lanes 16-31

  const float* xb = x   + (size_t)blockIdx.x * FEAT * NSYM;
  float*       ob = out + (size_t)blockIdx.x * FEAT * NSYM;

  // Warm L2 for the (shared, tiny) weight tables.
  __builtin_prefetch(wbT + (size_t)tid * FEAT, 0, 0);
  __builtin_prefetch(waT + (size_t)tid * FEAT, 0, 0);

  // ---------------- Stage 1: H[o', g] = sum_f WbT[o',f] * X[f,g] -----------
  for (int gp = 0; gp < NSYM; gp += 16) {
    // Cooperative load of X panel (transposed into Xs[g_local][f]), fp32->bf16.
    const float* src = xb + (size_t)tid * NSYM + gp;     // thread owns f = tid
    #pragma unroll
    for (int i = 0; i < 16; i += 4) {
      f32x4 v = *(const f32x4*)(src + i);
      Xs[(i + 0) * PS + tid] = f2bf(v.x);
      Xs[(i + 1) * PS + tid] = f2bf(v.y);
      Xs[(i + 2) * PS + tid] = f2bf(v.z);
      Xs[(i + 3) * PS + tid] = f2bf(v.w);
    }
    __syncthreads();

    #pragma unroll
    for (int t = 0; t < 2; ++t) {                        // 2 o'-tiles per wave
      const int orow = (wave * 2 + t) * 16;
      v8f acc = {};
      const unsigned short* arow = wbT + (size_t)(orow + l15) * FEAT;
      const unsigned short* brow = Xs + l15 * PS;
      #pragma unroll
      for (int ks = 0; ks < 8; ++ks) {                   // contract f: 8 x K=32
        Frag a, bf;
        const int kb = ks * 32 + lh * 8;
        a.q[0] = *(const u32x4*)(arow + kb);
        a.q[1] = *(const u32x4*)(arow + kb + 16);
        const int fb = ks * 32 + lh * 8;
        bf.q[0] = *(const u32x4*)(brow + fb);
        bf.q[1] = *(const u32x4*)(brow + fb + 16);
        acc = __builtin_amdgcn_wmma_f32_16x16x32_bf16(false, a.v, false, bf.v,
                                                      (short)0, acc, false, false);
      }
      #pragma unroll
      for (int r = 0; r < 8; ++r)                        // D -> Hs (bf16)
        Hs[(orow + lh * 8 + r) * HS + gp + l15] = f2bf(acc[r]);
    }
    __syncthreads();
  }

  // ------------- Stage 2+3 over k panels of 16 columns ---------------------
  const unsigned short* kfh = kfB + (size_t)wave * NSYM * NSYM;  // wave w == head w
  for (int kp = 0; kp < NSYM; kp += 16) {
    // Stage 2: Z[o', k] = sum_g H[o',g] * kfull[h][k][g]
    #pragma unroll
    for (int t = 0; t < 2; ++t) {
      const int orow = (wave * 2 + t) * 16;              // head = orow/32 = wave
      v8f acc = {};
      const unsigned short* arow = Hs + (size_t)(orow + l15) * HS;
      const unsigned short* brow = kfh + (size_t)(kp + l15) * NSYM;
      #pragma unroll
      for (int gs = 0; gs < 8; ++gs) {
        Frag a, bf;
        const int kb = gs * 32 + lh * 8;
        a.q[0] = *(const u32x4*)(arow + kb);
        a.q[1] = *(const u32x4*)(arow + kb + 16);
        const int gb = gs * 32 + lh * 8;
        bf.q[0] = *(const u32x4*)(brow + gb);
        bf.q[1] = *(const u32x4*)(brow + gb + 16);
        acc = __builtin_amdgcn_wmma_f32_16x16x32_bf16(false, a.v, false, bf.v,
                                                      (short)0, acc, false, false);
      }
      #pragma unroll
      for (int r = 0; r < 8; r += 2) {                   // packed cvt -> b32 store
        const int oc = orow + lh * 8 + r;
        *(unsigned*)&Zs[l15 * PS + oc] = pack2bf(acc[r], acc[r + 1]);
      }
    }
    __syncthreads();

    // Stage 3: Out[o2, k] = sum_o' WaT[o2,o'] * Z[o',k]  (+ bias)
    #pragma unroll
    for (int t = 0; t < 2; ++t) {
      const int orow = (wave * 2 + t) * 16;              // o2 tile base
      v8f acc = {};
      const unsigned short* arow = waT + (size_t)(orow + l15) * FEAT;
      const unsigned short* brow = Zs + l15 * PS;
      #pragma unroll
      for (int os = 0; os < 8; ++os) {
        Frag a, bf;
        const int kb = os * 32 + lh * 8;
        a.q[0] = *(const u32x4*)(arow + kb);
        a.q[1] = *(const u32x4*)(arow + kb + 16);
        const int obk = os * 32 + lh * 8;
        bf.q[0] = *(const u32x4*)(brow + obk);
        bf.q[1] = *(const u32x4*)(brow + obk + 16);
        acc = __builtin_amdgcn_wmma_f32_16x16x32_bf16(false, a.v, false, bf.v,
                                                      (short)0, acc, false, false);
      }
      #pragma unroll
      for (int r = 0; r < 8; ++r) {
        const int row = orow + lh * 8 + r;
        ob[(size_t)row * NSYM + kp + l15] = acc[r] + bias[row];
      }
    }
    __syncthreads();
  }
}

// ---------------------------------------------------------------------------
extern "C" void kernel_launch(void* const* d_in, const int* in_sizes, int n_in,
                              void* d_out, int out_size, void* d_ws, size_t ws_size,
                              hipStream_t stream) {
  (void)n_in; (void)out_size; (void)ws_size;
  const float* x    = (const float*)d_in[0];
  const float* wb   = (const float*)d_in[1];
  const float* kr   = (const float*)d_in[2];
  const float* wa   = (const float*)d_in[3];
  const float* bias = (const float*)d_in[4];
  const int*   pt   = (const int*)d_in[5];
  float*       outp = (float*)d_out;

  // Workspace layout: kfB (1 MB) | wbT (128 KB) | waT (128 KB)
  unsigned short* kfB = (unsigned short*)d_ws;
  unsigned short* wbT = (unsigned short*)((char*)d_ws +
                        (size_t)NHEAD * NSYM * NSYM * sizeof(unsigned short));
  unsigned short* waT = wbT + FEAT * FEAT;

  const int nb = in_sizes[0] / (FEAT * NSYM);            // batch = 1024

  prep_kernel<<<(NHEAD * NSYM * NSYM) / 256, 256, 0, stream>>>(
      wb, kr, wa, pt, kfB, wbT, waT);
  fused_kernel<<<nb, 256, 0, stream>>>(x, wbT, kfB, waT, bias, outp);
}
